// CopyMB_48876727828995
// MI455X (gfx1250) — compile-verified, matrix operand order checked by reading, not run on hardware
//
#include <hip/hip_runtime.h>

#define BATCH   8
#define TLEN    100
#define HDIM    256
#define DDIM    288
#define ROWS    800
#define NSTEP   11

typedef __attribute__((ext_vector_type(16))) __bf16 v16bf;
typedef __attribute__((ext_vector_type(8)))  __bf16 v8bf;
typedef __attribute__((ext_vector_type(8)))  float  v8f;

__device__ __forceinline__ __bf16 f2bf(float x) {
  union { float f; unsigned u; } c; c.f = x;
  unsigned r = (c.u + 0x7FFFu + ((c.u >> 16) & 1u)) >> 16;
  union { unsigned short s; __bf16 b; } o; o.s = (unsigned short)r;
  return o.b;
}
__device__ __forceinline__ float sigmf(float x) { return 1.0f / (1.0f + __expf(-x)); }

// 16-bf16 fragment: {p[0..7], p[16..23]} (caller biases p by +8 for lanes>=16)
__device__ __forceinline__ v16bf load_frag_bf(const __bf16* p) {
  union { v16bf v; v8bf h[2]; } f;
  f.h[0] = *(const v8bf*)(p);
  f.h[1] = *(const v8bf*)(p + 16);
  return f.v;
}
__device__ __forceinline__ v16bf zero_frag() {
  union { v16bf v; unsigned u[8]; } f;
#pragma unroll
  for (int i = 0; i < 8; ++i) f.u[i] = 0u;
  return f.v;
}

// ---- GEMM: C[M,N] = A[M,K] @ B[N,K]^T + bias. One wave per 16x64 strip.
// K compile-time (fully unrolled); NG=true adds N-edge guards (N=3/51 cases).
// M assumed multiple of 16. Cf/Cb: optional f32 / bf16 outputs.
template <int K, bool NG>
__global__ void gemm_bf16_kernel(const __bf16* __restrict__ A, int lda,
                                 const __bf16* __restrict__ B, int ldb,
                                 float* __restrict__ Cf, __bf16* __restrict__ Cb,
                                 int ldc, const float* __restrict__ bias,
                                 int M, int N) {
  const int lane = threadIdx.x & 31;
  const int n0   = blockIdx.x * 64;
  const int m0   = blockIdx.y * 16;
  const int kb   = (lane >> 4) * 8;
  const int l15  = lane & 15;
  const __bf16* ap  = A + (long)(m0 + l15) * lda + kb;
  const __bf16* bp0 = B + (long)(n0 + l15) * ldb + kb;
  v8f acc0 = {}, acc1 = {}, acc2 = {}, acc3 = {};
#pragma unroll
  for (int k0 = 0; k0 < K; k0 += 32) {
    if (k0 + 32 < K) __builtin_prefetch(ap + k0 + 32, 0, 3);
    v16bf a = load_frag_bf(ap + k0);
    v16bf b0, b1, b2, b3;
    if (NG) {
      b0 = (n0 + l15      < N) ? load_frag_bf(bp0 + k0)                  : zero_frag();
      b1 = (n0 + 16 + l15 < N) ? load_frag_bf(bp0 + (long)16 * ldb + k0) : zero_frag();
      b2 = (n0 + 32 + l15 < N) ? load_frag_bf(bp0 + (long)32 * ldb + k0) : zero_frag();
      b3 = (n0 + 48 + l15 < N) ? load_frag_bf(bp0 + (long)48 * ldb + k0) : zero_frag();
    } else {
      b0 = load_frag_bf(bp0 + k0);
      b1 = load_frag_bf(bp0 + (long)16 * ldb + k0);
      b2 = load_frag_bf(bp0 + (long)32 * ldb + k0);
      b3 = load_frag_bf(bp0 + (long)48 * ldb + k0);
    }
    acc0 = __builtin_amdgcn_wmma_f32_16x16x32_bf16(false, a, false, b0, (short)0, acc0, false, false);
    acc1 = __builtin_amdgcn_wmma_f32_16x16x32_bf16(false, a, false, b1, (short)0, acc1, false, false);
    acc2 = __builtin_amdgcn_wmma_f32_16x16x32_bf16(false, a, false, b2, (short)0, acc2, false, false);
    acc3 = __builtin_amdgcn_wmma_f32_16x16x32_bf16(false, a, false, b3, (short)0, acc3, false, false);
  }
  const int rbase = m0 + ((lane >> 4) ? 8 : 0);
  v8f accs[4] = {acc0, acc1, acc2, acc3};
#pragma unroll
  for (int t = 0; t < 4; ++t) {
    int col = n0 + t * 16 + l15;
    if (NG && col >= N) continue;
    float bv = bias ? bias[col] : 0.0f;
#pragma unroll
    for (int i = 0; i < 8; ++i) {
      float val = accs[t][i] + bv;
      long  idx = (long)(rbase + i) * ldc + col;
      if (Cf) Cf[idx] = val;
      if (Cb) Cb[idx] = f2bf(val);
    }
  }
}

// ---- converters ----
__global__ void cvt_kernel(const float* __restrict__ s, __bf16* __restrict__ d, int n, int relu) {
  int i = blockIdx.x * 256 + threadIdx.x;
  if (i < n) { float x = s[i]; if (relu) x = fmaxf(x, 0.0f); d[i] = f2bf(x); }
}
__global__ void cvt_slice_kernel(const float* __restrict__ s, int srow, int col0,
                                 __bf16* __restrict__ d, int rows, int cols) {
  int i = blockIdx.x * 256 + threadIdx.x;
  if (i >= rows * cols) return;
  int r = i / cols, c = i % cols;
  d[i] = f2bf(s[(long)r * srow + col0 + c]);
}

// ---- embedding gather -> bf16 (only used as GEMM A) ----
__global__ void embed_kernel(const float* __restrict__ we, const int* __restrict__ tok,
                             __bf16* __restrict__ emb) {
  int r = blockIdx.x, c = threadIdx.x;
  emb[(long)r * HDIM + c] = f2bf(we[(long)tok[r] * HDIM + c]);
}

// ---- persistent encoder GRU scan ----
// h kept as: bf16 in LDS (ping-pong, A-fragments), f32 in registers (same-lane reuse:
// the gate-combine lane mapping equals the C/D fragment layout every step).
// An opaque zero OFFSET (not the pointer) is laundered through inline asm each step:
// this keeps the Whh loads in-loop (no LICM hoist + scratch spill) while preserving
// pointer provenance, so they lower as global_load_b128 (LOADcnt only), independent
// of the ds_load_b128 A-fragment stream (DScnt).
__global__ void enc_gru_kernel(const float* __restrict__ GI, const __bf16* __restrict__ Whh,
                               const float* __restrict__ bhh, float* __restrict__ o, int dir) {
  __shared__ __align__(16) __bf16 hb[2][16 * HDIM];
  for (int idx = threadIdx.x; idx < 2 * 16 * HDIM; idx += blockDim.x)
    (&hb[0][0])[idx] = f2bf(0.0f);
  __syncthreads();
  const int wave   = threadIdx.x >> 5;
  const int lane   = threadIdx.x & 31;
  const int l15    = lane & 15;
  const int kb     = (lane >> 4) * 8;
  const int rowoff = (lane >> 4) * 8;
  const bool active = (lane < 16);   // lanes 0..15 own batch rows 0..7; 16..31 own pad rows
  long boff[6];
#pragma unroll
  for (int ct = 0; ct < 2; ++ct)
#pragma unroll
    for (int g = 0; g < 3; ++g)
      boff[ct * 3 + g] = (long)(g * HDIM + wave * 32 + ct * 16 + l15) * HDIM + kb;
  float hreg[2][8];
#pragma unroll
  for (int ct = 0; ct < 2; ++ct)
#pragma unroll
    for (int i = 0; i < 8; ++i) hreg[ct][i] = 0.0f;
  int cur = 0;
  for (int s = 0; s < TLEN; ++s) {
    const int tsrc = dir ? (TLEN - 1 - s) : s;
    long ofs = 0;
    asm volatile("" : "+s"(ofs));         // opaque per-iteration zero: defeats LICM only
    const __bf16* wp = Whh + ofs;
    const __bf16* hbc = &hb[cur][0] + l15 * HDIM + kb;
    v8f acc[6] = {};
#pragma unroll
    for (int k0 = 0; k0 < HDIM; k0 += 32) {
      v16bf a = load_frag_bf(hbc + k0);
#pragma unroll
      for (int t = 0; t < 6; ++t) {
        v16bf b = load_frag_bf(wp + boff[t] + k0);
        acc[t] = __builtin_amdgcn_wmma_f32_16x16x32_bf16(false, a, false, b, (short)0, acc[t], false, false);
      }
    }
    __bf16* hbn = &hb[cur ^ 1][0];
#pragma unroll
    for (int ct = 0; ct < 2; ++ct) {
      int c = wave * 32 + ct * 16 + l15;
#pragma unroll
      for (int i = 0; i < 8; ++i) {
        float hnew = 0.0f;
        if (active) {
          int row = i;                    // batch row
          const float* gi = GI + (long)(row * TLEN + tsrc) * (3 * HDIM);
          float rr = sigmf(gi[c]            + acc[ct * 3 + 0][i] + bhh[c]);
          float zz = sigmf(gi[HDIM + c]     + acc[ct * 3 + 1][i] + bhh[HDIM + c]);
          float nn = tanhf(gi[2 * HDIM + c] + rr * (acc[ct * 3 + 2][i] + bhh[2 * HDIM + c]));
          hnew = (1.0f - zz) * nn + zz * hreg[ct][i];
          o[(long)(row * TLEN + tsrc) * HDIM + c] = hnew;
        }
        hreg[ct][i] = hnew;
        hbn[(rowoff + i) * HDIM + c] = f2bf(hnew);
      }
    }
    __syncthreads();
    cur ^= 1;
  }
}

// ---- o = 0.5*(of+ob): write f32 + bf16 ----
__global__ void avg_kernel(const float* a, const float* b, float* c, __bf16* cb, int n) {
  int i = blockIdx.x * blockDim.x + threadIdx.x;
  if (i < n) { float v = 0.5f * (a[i] + b[i]); c[i] = v; cb[i] = f2bf(v); }
}

// ---- CRF (3 tags), lanes 0..7 run one sequence each ----
__global__ void crf_kernel(const float* __restrict__ emi, const int* __restrict__ tags,
                           const int* __restrict__ tokens, const float* __restrict__ start,
                           const float* __restrict__ endp, const float* __restrict__ trans,
                           float* __restrict__ out_crf) {
  __shared__ float vals[BATCH];
  int b = threadIdx.x;
  if (b < BATCH) {
    int   t0    = tags[b * TLEN];
    float score = start[t0] + emi[(long)(b * TLEN) * 3 + t0];
    int   prev  = t0;
    float alpha[3];
    for (int j = 0; j < 3; ++j) alpha[j] = start[j] + emi[(long)(b * TLEN) * 3 + j];
    for (int t = 1; t < TLEN; ++t) {
      float m  = (tokens[b * TLEN + t] != 0) ? 1.0f : 0.0f;
      int   tg = tags[b * TLEN + t];
      const float* e = emi + (long)(b * TLEN + t) * 3;
      score += (trans[prev * 3 + tg] + e[tg]) * m;
      if (m > 0.0f) prev = tg;
      float na[3];
      for (int j = 0; j < 3; ++j) {
        float m0 = alpha[0] + trans[0 * 3 + j];
        float m1 = alpha[1] + trans[1 * 3 + j];
        float m2 = alpha[2] + trans[2 * 3 + j];
        float mx = fmaxf(m0, fmaxf(m1, m2));
        na[j] = mx + __logf(__expf(m0 - mx) + __expf(m1 - mx) + __expf(m2 - mx)) + e[j];
      }
      if (m > 0.0f) { alpha[0] = na[0]; alpha[1] = na[1]; alpha[2] = na[2]; }
    }
    float num = score + endp[prev];
    float a0 = alpha[0] + endp[0], a1 = alpha[1] + endp[1], a2 = alpha[2] + endp[2];
    float mx = fmaxf(a0, fmaxf(a1, a2));
    float den = mx + __logf(__expf(a0 - mx) + __expf(a1 - mx) + __expf(a2 - mx));
    vals[b] = num - den;
  }
  __syncthreads();
  if (threadIdx.x == 0) {
    float s = 0.0f;
    for (int i = 0; i < BATCH; ++i) s += vals[i];
    *out_crf = -(s / (float)BATCH);
  }
}

// ---- cat_o = [o, bio_emb[bio_gold]] (f32) + relu-bf16 copy ----
__global__ void cat_build_kernel(const float* __restrict__ o, const float* __restrict__ bio_emb,
                                 const int* __restrict__ bio_gold, float* __restrict__ cat_o,
                                 __bf16* __restrict__ cat_rbf) {
  int idx = blockIdx.x * blockDim.x + threadIdx.x;
  if (idx >= ROWS * DDIM) return;
  int r = idx / DDIM, c = idx % DDIM;
  float v = (c < HDIM) ? o[(long)r * HDIM + c] : bio_emb[bio_gold[r] * 32 + (c - HDIM)];
  cat_o[idx]   = v;
  cat_rbf[idx] = f2bf(fmaxf(v, 0.0f));
}

// ---- h0 = o[b, len_b-1] (f32 + bf16 + relu-bf16) ----
__global__ void init_h_kernel(const float* __restrict__ o, const int* __restrict__ lengths,
                              float* __restrict__ hf, __bf16* __restrict__ hb,
                              __bf16* __restrict__ hr) {
  int r = blockIdx.x, c = threadIdx.x;
  int b = r / TLEN;
  int t = lengths[b] - 1;
  float v = o[(long)(b * TLEN + t) * HDIM + c];
  long  i = (long)r * HDIM + c;
  hf[i] = v; hb[i] = f2bf(v); hr[i] = f2bf(fmaxf(v, 0.0f));
}

// ---- GRU gate combine; writes h as f32 + bf16 + relu-bf16 ----
__global__ void gru_combine_kernel(const float* __restrict__ gi, const float* __restrict__ gh,
                                   const float* __restrict__ h, float* __restrict__ hf,
                                   __bf16* __restrict__ hb, __bf16* __restrict__ hr) {
  int idx = blockIdx.x * blockDim.x + threadIdx.x;
  if (idx >= ROWS * HDIM) return;
  int r = idx >> 8, c = idx & 255;
  const float* gir = gi + (long)r * 3 * HDIM;
  const float* ghr = gh + (long)r * 3 * HDIM;
  float rr = sigmf(gir[c] + ghr[c]);
  float zz = sigmf(gir[HDIM + c] + ghr[HDIM + c]);
  float nn = tanhf(gir[2 * HDIM + c] + rr * ghr[2 * HDIM + c]);
  float v  = (1.0f - zz) * nn + zz * h[idx];
  hf[idx] = v; hb[idx] = f2bf(v); hr[idx] = f2bf(fmaxf(v, 0.0f));
}

// ---- odd-step attention: wave w owns j = w, w+8, ... ----
__global__ void attn_kernel(const float* __restrict__ hq, const float* __restrict__ Kbuf,
                            const float* __restrict__ e2w, const float* __restrict__ e2b,
                            float* __restrict__ logits) {
  int r = blockIdx.x, tid = threadIdx.x, wave = tid >> 5, lane = tid & 31;
  __shared__ float hs[HDIM], ws[HDIM];
  hs[tid] = hq[(long)r * HDIM + tid];
  ws[tid] = e2w[tid];
  __syncthreads();
  const float* Kr   = Kbuf + (long)((r & 7) * TLEN) * HDIM;
  float        e2bv = e2b[0];
  for (int j = wave; j < TLEN; j += 8) {
    const float* Kj = Kr + (long)j * HDIM;
    float v = 0.0f;
#pragma unroll
    for (int c0 = 0; c0 < HDIM; c0 += 32) {
      int c = c0 + lane;
      v += fmaxf(hs[c] + Kj[c], 0.0f) * ws[c];
    }
    for (int off = 16; off; off >>= 1) v += __shfl_down(v, off);
    if (lane == 0) logits[(long)r * TLEN + j] = v + e2bv;
  }
}

// ---- masked log-softmax CE, atomically accumulated ----
__global__ void ce_kernel(const float* __restrict__ logits, int ncls, int stride,
                          const int* __restrict__ seq_gold, const int* __restrict__ maskd,
                          int step, float* __restrict__ acc) {
  int r = blockIdx.x, tid = threadIdx.x;
  const float* lr = logits + (long)r * stride;
  __shared__ float redm[4], reds[4];
  float mx = -1e30f;
  for (int j = tid; j < ncls; j += blockDim.x) mx = fmaxf(mx, lr[j]);
  for (int off = 16; off; off >>= 1) mx = fmaxf(mx, __shfl_down(mx, off));
  if ((tid & 31) == 0) redm[tid >> 5] = mx;
  __syncthreads();
  if (tid == 0) {
    float m = redm[0];
    for (int w = 1; w < 4; ++w) m = fmaxf(m, redm[w]);
    redm[0] = m;
  }
  __syncthreads();
  mx = redm[0];
  float s = 0.0f;
  for (int j = tid; j < ncls; j += blockDim.x) s += __expf(lr[j] - mx);
  for (int off = 16; off; off >>= 1) s += __shfl_down(s, off);
  if ((tid & 31) == 0) reds[tid >> 5] = s;
  __syncthreads();
  if (tid == 0) {
    float ss = 0.0f;
    for (int w = 0; w < 4; ++w) ss += reds[w];
    int   g  = seq_gold[r * NSTEP + step];
    float ce = (mx + __logf(ss)) - lr[g];
    float m  = (float)maskd[r * NSTEP + step];
    atomicAdd(acc, ce * m);
  }
}

// ---- teacher forcing gathers (write bf16 GEMM inputs directly) ----
__global__ void gather_rel_kernel(const float* __restrict__ rel_emb, const int* __restrict__ seq_gold,
                                  int step, __bf16* __restrict__ dec_in) {
  int r = blockIdx.x, c = threadIdx.x;
  dec_in[(long)r * HDIM + c] = f2bf(rel_emb[(long)seq_gold[r * NSTEP + step] * HDIM + c]);
}
__global__ void gather_picked_kernel(const float* __restrict__ cat_o, const int* __restrict__ seq_gold,
                                     int step, __bf16* __restrict__ Abuf) {
  int idx = blockIdx.x * blockDim.x + threadIdx.x;
  if (idx >= ROWS * DDIM) return;
  int r = idx / DDIM, c = idx % DDIM;
  int g = seq_gold[r * NSTEP + step];
  Abuf[idx] = f2bf(fmaxf(cat_o[(long)((r & 7) * TLEN + g) * DDIM + c], 0.0f));
}

__global__ void mdf_sum_kernel(const int* __restrict__ md, float* __restrict__ out) {
  __shared__ float part[8];
  float s = 0.0f;
  for (int i = threadIdx.x; i < ROWS * NSTEP; i += 256) s += (float)md[i];
  for (int off = 16; off; off >>= 1) s += __shfl_down(s, off);
  if ((threadIdx.x & 31) == 0) part[threadIdx.x >> 5] = s;
  __syncthreads();
  if (threadIdx.x == 0) {
    float t = 0.0f;
    for (int w = 0; w < 8; ++w) t += part[w];
    out[0] = t;
  }
}

__global__ void zero1_kernel(float* p) { if (threadIdx.x == 0) p[0] = 0.0f; }

__global__ void finalize_kernel(const float* __restrict__ acc, float* __restrict__ out) {
  if (threadIdx.x == 0) {
    float d = acc[0] / acc[1];
    out[0] = acc[2] + d;
    out[1] = acc[2];
    out[2] = d;
  }
}

extern "C" void kernel_launch(void* const* d_in, const int* in_sizes, int n_in,
                              void* d_out, int out_size, void* d_ws, size_t ws_size,
                              hipStream_t stream) {
  (void)in_sizes; (void)n_in; (void)out_size; (void)ws_size;
  const int*   tokens   = (const int*)d_in[0];
  const int*   bio_gold = (const int*)d_in[1];
  const int*   seq_gold = (const int*)d_in[2];
  const int*   lengths  = (const int*)d_in[3];
  const int*   mask_d   = (const int*)d_in[4];
  const float* word_emb = (const float*)d_in[5];
  const float* rel_emb  = (const float*)d_in[6];
  const float* bio_emb  = (const float*)d_in[7];
  const float* fWih = (const float*)d_in[8],  *fWhh = (const float*)d_in[9];
  const float* fbih = (const float*)d_in[10], *fbhh = (const float*)d_in[11];
  const float* bWih = (const float*)d_in[12], *bWhh = (const float*)d_in[13];
  const float* bbih = (const float*)d_in[14], *bbhh = (const float*)d_in[15];
  const float* dWih = (const float*)d_in[16], *dWhh = (const float*)d_in[17];
  const float* dbih = (const float*)d_in[18], *dbhh = (const float*)d_in[19];
  const float* emi_W = (const float*)d_in[20], *emi_b = (const float*)d_in[21];
  const float* relb_W = (const float*)d_in[22], *relb_b = (const float*)d_in[23];
  const float* e1_W = (const float*)d_in[24], *e1_b = (const float*)d_in[25];
  const float* e2_W = (const float*)d_in[26], *e2_b = (const float*)d_in[27];
  const float* cat_W = (const float*)d_in[28], *cat_b = (const float*)d_in[29];
  const float* crf_start = (const float*)d_in[30];
  const float* crf_end   = (const float*)d_in[31];
  const float* crf_trans = (const float*)d_in[32];
  float* out = (float*)d_out;

  // ---- workspace: f32 region, then bf16 region ----
  float* F     = (float*)d_ws;
  float* o_f   = F;                  // 800*256
  float* o_b   = o_f   + 204800;
  float* o_av  = o_b   + 204800;
  float* cat_o = o_av  + 204800;     // 800*288
  float* Kbuf  = cat_o + 230400;
  float* hA_f  = Kbuf  + 204800;
  float* hB_f  = hA_f  + 204800;
  float* hq    = hB_f  + 204800;
  float* logit = hq    + 204800;     // 800*100
  float* emi   = logit + 80000;      // 800*3
  float* gi_f  = emi   + 2400;       // 800*768 (enc GI_f, then dec gi)
  float* gh_f  = gi_f  + 614400;     // 800*768 (enc GI_b, then dec gh)
  float* accum = gh_f  + 614400;     // [0]=dec_sum [1]=mdf_sum [2]=crf

  __bf16* Bp      = (__bf16*)(accum + 16);
  __bf16* emb_bf  = Bp;               // 800*256
  __bf16* o_bf    = emb_bf  + 204800;
  __bf16* cat_rbf = o_bf    + 204800; // 800*288
  __bf16* hA_bf   = cat_rbf + 230400;
  __bf16* hB_bf   = hA_bf   + 204800;
  __bf16* hA_rbf  = hB_bf   + 204800;
  __bf16* hB_rbf  = hA_rbf  + 204800;
  __bf16* din_bf  = hB_rbf  + 204800; // 800*256
  __bf16* Abuf_bf = din_bf  + 204800; // 800*288
  __bf16* fWih_bf = Abuf_bf + 230400; // 768*256
  __bf16* bWih_bf = fWih_bf + 196608;
  __bf16* fWhh_bf = bWih_bf + 196608;
  __bf16* bWhh_bf = fWhh_bf + 196608;
  __bf16* dWih_bf = bWhh_bf + 196608;
  __bf16* dWhh_bf = dWih_bf + 196608;
  __bf16* emiW_bf = dWhh_bf + 196608; // 3*256
  __bf16* relW_bf = emiW_bf + 768;    // 51*256
  __bf16* catW_bf = relW_bf + 13056;  // 256*288
  __bf16* Wh_bf   = catW_bf + 73728;  // 256*256 (e1_W[:, :256])
  __bf16* Wc_bf   = Wh_bf   + 65536;  // 256*288 (e1_W[:, 256:])

  dim3 wv(32);
  auto CVT = [&](const float* s, __bf16* d, int n) {
    cvt_kernel<<<(n + 255) / 256, 256, 0, stream>>>(s, d, n, 0);
  };

  // ---- weight conversion (once per launch) ----
  CVT(fWih, fWih_bf, 196608); CVT(bWih, bWih_bf, 196608);
  CVT(fWhh, fWhh_bf, 196608); CVT(bWhh, bWhh_bf, 196608);
  CVT(dWih, dWih_bf, 196608); CVT(dWhh, dWhh_bf, 196608);
  CVT(emi_W, emiW_bf, 768);   CVT(relb_W, relW_bf, 13056);
  CVT(cat_W, catW_bf, 73728);
  cvt_slice_kernel<<<(65536 + 255) / 256, 256, 0, stream>>>(e1_W, 544, 0,   Wh_bf, 256, 256);
  cvt_slice_kernel<<<(73728 + 255) / 256, 256, 0, stream>>>(e1_W, 544, 256, Wc_bf, 256, 288);

  // ---- encoder ----
  embed_kernel<<<ROWS, HDIM, 0, stream>>>(word_emb, tokens, emb_bf);
  gemm_bf16_kernel<256, false><<<dim3(12, 50), wv, 0, stream>>>(emb_bf, 256, fWih_bf, 256, gi_f, nullptr, 768, fbih, ROWS, 768);
  gemm_bf16_kernel<256, false><<<dim3(12, 50), wv, 0, stream>>>(emb_bf, 256, bWih_bf, 256, gh_f, nullptr, 768, bbih, ROWS, 768);
  enc_gru_kernel<<<1, 256, 0, stream>>>(gi_f, fWhh_bf, fbhh, o_f, 0);
  enc_gru_kernel<<<1, 256, 0, stream>>>(gh_f, bWhh_bf, bbhh, o_b, 1);
  avg_kernel<<<(ROWS * HDIM + 255) / 256, 256, 0, stream>>>(o_f, o_b, o_av, o_bf, ROWS * HDIM);

  // ---- CRF ----
  gemm_bf16_kernel<256, true><<<dim3(1, 50), wv, 0, stream>>>(o_bf, 256, emiW_bf, 256, emi, nullptr, 3, emi_b, ROWS, 3);
  crf_kernel<<<1, 32, 0, stream>>>(emi, bio_gold, tokens, crf_start, crf_end, crf_trans, accum + 2);

  // ---- decoder setup ----
  cat_build_kernel<<<(ROWS * DDIM + 255) / 256, 256, 0, stream>>>(o_av, bio_emb, bio_gold, cat_o, cat_rbf);
  init_h_kernel<<<ROWS, HDIM, 0, stream>>>(o_av, lengths, hA_f, hA_bf, hA_rbf);
  gemm_bf16_kernel<288, false><<<dim3(4, 50), wv, 0, stream>>>(cat_rbf, 288, catW_bf, 288, nullptr, din_bf, 256, cat_b, ROWS, 256);
  gemm_bf16_kernel<288, false><<<dim3(4, 50), wv, 0, stream>>>(cat_rbf, 288, Wc_bf, 288, Kbuf, nullptr, 256, e1_b, ROWS, 256);
  zero1_kernel<<<1, 32, 0, stream>>>(accum);
  mdf_sum_kernel<<<1, 256, 0, stream>>>(mask_d, accum + 1);

  // ---- decoder steps ----
  float  *hcf = hA_f,  *hnf = hB_f;
  __bf16 *hcb = hA_bf, *hnb = hB_bf, *hcr = hA_rbf, *hnr = hB_rbf;
  for (int i = 0; i < NSTEP; ++i) {
    gemm_bf16_kernel<256, false><<<dim3(12, 50), wv, 0, stream>>>(din_bf, 256, dWih_bf, 256, gi_f, nullptr, 768, dbih, ROWS, 768);
    gemm_bf16_kernel<256, false><<<dim3(12, 50), wv, 0, stream>>>(hcb,    256, dWhh_bf, 256, gh_f, nullptr, 768, dbhh, ROWS, 768);
    gru_combine_kernel<<<(ROWS * HDIM + 255) / 256, 256, 0, stream>>>(gi_f, gh_f, hcf, hnf, hnb, hnr);
    { float*  t = hcf; hcf = hnf; hnf = t; }
    { __bf16* t = hcb; hcb = hnb; hnb = t; }
    { __bf16* t = hcr; hcr = hnr; hnr = t; }
    if ((i & 1) == 0) {
      gemm_bf16_kernel<256, true><<<dim3(1, 50), wv, 0, stream>>>(hcb, 256, relW_bf, 256, logit, nullptr, 51, relb_b, ROWS, 51);
      ce_kernel<<<ROWS, 128, 0, stream>>>(logit, 51, 51, seq_gold, mask_d, i, accum);
      if (i < NSTEP - 1)
        gather_rel_kernel<<<ROWS, HDIM, 0, stream>>>(rel_emb, seq_gold, i, din_bf);
    } else {
      gemm_bf16_kernel<256, false><<<dim3(4, 50), wv, 0, stream>>>(hcr, 256, Wh_bf, 256, hq, nullptr, 256, nullptr, ROWS, 256);
      attn_kernel<<<ROWS, HDIM, 0, stream>>>(hq, Kbuf, e2_W, e2_b, logit);
      ce_kernel<<<ROWS, 128, 0, stream>>>(logit, TLEN, TLEN, seq_gold, mask_d, i, accum);
      gather_picked_kernel<<<(ROWS * DDIM + 255) / 256, 256, 0, stream>>>(cat_o, seq_gold, i, Abuf_bf);
      gemm_bf16_kernel<288, false><<<dim3(4, 50), wv, 0, stream>>>(Abuf_bf, 288, catW_bf, 288, nullptr, din_bf, 256, cat_b, ROWS, 256);
    }
  }
  finalize_kernel<<<1, 32, 0, stream>>>(accum, out);
}
